// SEMultitask_42588895707908
// MI455X (gfx1250) — matrix operand order repaired and checked by reading, not compiled, not run on hardware
//
#include <hip/hip_runtime.h>
#include <cmath>

// ---------------------------------------------------------------------------
// MI455X (gfx1250) implementation: all GEMMs via v_wmma_f32_16x16x32_bf16
// (wave32, fp32 accumulate), 2x2 register blocking (4 WMMA / wave / k-step),
// double-buffered LDS staged with GLOBAL_LOAD_ASYNC_TO_LDS_B128 (ASYNCcnt).
// Workload is matrix-bound (~0.25 TFLOP, working set < L2), so maximize
// matrix-pipe issue density and use the CDNA5 async LDS-DMA path.
// ---------------------------------------------------------------------------

typedef __attribute__((ext_vector_type(16))) __bf16 v16bf;
typedef __attribute__((ext_vector_type(8)))  float  v8f;

union FragU { v16bf v; unsigned u[8]; };

#define Bn    16
#define Tn    512
#define Dn    300
#define DPn   320
#define Hn    4
#define DKn   75
#define DKPn  128      // padded head dim (multiple of 64 for the N-tile)
#define HIDn  512
#define ROWSn (Bn*Tn)  // 8192
#define BHn   (Bn*Hn)  // 64
#define NEGV  (-1e9f)

__device__ __forceinline__ unsigned lds_off32(const void* p) {
  // generic pointer to LDS: low 32 bits carry the LDS byte offset
  return (unsigned)(size_t)p;
}
__device__ __forceinline__ void async_ld_b128(unsigned ldsOff, const void* gp) {
  asm volatile("global_load_async_to_lds_b128 %0, %1, off"
               :: "v"(ldsOff), "v"((unsigned long long)(size_t)gp) : "memory");
}
__device__ __forceinline__ void wait_async0() {
  asm volatile("s_wait_asynccnt 0x0" ::: "memory");
}

// ---------------------------------------------------------------------------
// fp32 -> bf16 convert with zero padding (weights + embeddings)
// ---------------------------------------------------------------------------
__global__ void cvt_pad_kernel(const float* __restrict__ src, int R, int C, int lds,
                               __bf16* __restrict__ dst, int Rp, int Cp, int ldd) {
  long idx = (long)blockIdx.x * 256 + threadIdx.x;
  long tot = (long)Rp * Cp;
  if (idx >= tot) return;
  int r = (int)(idx / Cp), c = (int)(idx % Cp);
  float v = (r < R && c < C) ? src[(long)r * lds + c] : 0.0f;
  dst[(long)r * ldd + c] = (__bf16)v;
}

// zero the dk-padding columns [DKn, DKPn) of a head buffer [BHn*Tn, DKPn]
__global__ void zero_headpad_kernel(__bf16* __restrict__ q) {
  const int padw = DKPn - DKn;
  long idx = (long)blockIdx.x * 256 + threadIdx.x;
  long tot = (long)BHn * Tn * padw;
  if (idx >= tot) return;
  long r = idx / padw;
  int  c = DKn + (int)(idx % padw);
  q[r * DKPn + c] = (__bf16)0.0f;
}

// ---------------------------------------------------------------------------
// Generic bf16 WMMA GEMM: C = act(scale * A@B + bias), batched over grid.z.
// Block tile 128x64 (8 waves); each wave 32x32 = 2x2 WMMA subtiles.
//   bmode: 0 -> B is KxN row-major (weights, LDS-transposed via ds_store)
//          1 -> B is NxK row-major (K^T path, async B128 straight in)
//   mode : 0 bf16 store | 1 fp32 store | 2 fp32 += (residual)
//          3 head-split bf16 (proj -> [B*H,T,DKPn]) | 4 head-merge bf16
//   act  : 0 none | 1 gelu | 2 sigmoid
// ---------------------------------------------------------------------------
__global__ __launch_bounds__(256)
void gemm_kernel(const __bf16* __restrict__ A, int lda, long sA,
                 const __bf16* __restrict__ Bm, int ldb, long sB, int bmode,
                 const float* __restrict__ bias,
                 float* Cf, __bf16* Cb, int ldc, long sC,
                 int Kp, int Mreal, int Nreal,
                 int mode, int act, float scale) {
  __shared__ __bf16 As[2][128][40];   // stride 40 elems = 80B (16B aligned rows)
  __shared__ __bf16 Bs[2][64][40];    // stored transposed: Bs[n][k]

  const int tid  = threadIdx.x;
  const int wave = tid >> 5, lane = tid & 31;
  const int wm = wave >> 1, wn = wave & 1;         // 4x2 wave grid -> 128x64
  const long z = blockIdx.z;
  const int nBlock = blockIdx.x * 64;
  const int mBlock = blockIdx.y * 128;

  const __bf16* Ap = A + z * sA + (long)mBlock * lda;
  const __bf16* Bp = Bm + z * sB;

  auto stage = [&](int buf, int k0) {
    // A tile 128x32: 512 x 16B async DMA chunks, 2 per thread
#pragma unroll
    for (int i = 0; i < 2; ++i) {
      int idx = tid + i * 256;
      int r = idx >> 2, ch = (idx & 3) * 8;
      async_ld_b128(lds_off32(&As[buf][r][ch]), Ap + (long)r * lda + k0 + ch);
    }
    if (bmode == 1) {
      // B rows are k-contiguous -> async straight into Bs[n][k]
      int n = tid >> 2, ch = (tid & 3) * 8;
      async_ld_b128(lds_off32(&Bs[buf][n][ch]),
                    Bp + (long)(nBlock + n) * ldb + k0 + ch);
    } else {
      // weights KxN: transpose while staging
#pragma unroll
      for (int i = 0; i < 4; ++i) {
        int idx = tid + i * 256;
        int k = idx >> 5;
        int n = (idx & 31) * 2;
        unsigned u = *(const unsigned*)(Bp + (long)(k0 + k) * ldb + nBlock + n);
        Bs[buf][n][k]     = ((const __bf16*)&u)[0];
        Bs[buf][n + 1][k] = ((const __bf16*)&u)[1];
      }
    }
  };

  v8f zero8 = {0.f, 0.f, 0.f, 0.f, 0.f, 0.f, 0.f, 0.f};
  v8f accs[2][2] = {{zero8, zero8}, {zero8, zero8}};

  const int nk = Kp >> 5;
  const int half = lane >> 4;
  const int lm = lane & 15;

  stage(0, 0);
  for (int kt = 0; kt < nk; ++kt) {
    wait_async0();            // this wave's async DMA into buf kt&1 done
    __syncthreads();          // everyone's staging of buf kt&1 visible
    if (kt + 1 < nk) stage((kt + 1) & 1, (kt + 1) << 5);
    const int buf = kt & 1;

    FragU fa0, fa1, fb0, fb1;
#pragma unroll
    for (int i = 0; i < 8; ++i) {
      int ka = 2 * i + 16 * half;                                 // A 16x32 MxK
      int kb = 2 * i + 16 * half;                                 // B 32x16 KxN
      fa0.u[i] = *(const unsigned*)&As[buf][wm * 32 + lm][ka];
      fa1.u[i] = *(const unsigned*)&As[buf][wm * 32 + 16 + lm][ka];
      fb0.u[i] = *(const unsigned*)&Bs[buf][wn * 32 + lm][kb];
      fb1.u[i] = *(const unsigned*)&Bs[buf][wn * 32 + 16 + lm][kb];
    }
    accs[0][0] = __builtin_amdgcn_wmma_f32_16x16x32_bf16(
        false, fa0.v, false, fb0.v, (short)0, accs[0][0], false, false);
    accs[0][1] = __builtin_amdgcn_wmma_f32_16x16x32_bf16(
        false, fa0.v, false, fb1.v, (short)0, accs[0][1], false, false);
    accs[1][0] = __builtin_amdgcn_wmma_f32_16x16x32_bf16(
        false, fa1.v, false, fb0.v, (short)0, accs[1][0], false, false);
    accs[1][1] = __builtin_amdgcn_wmma_f32_16x16x32_bf16(
        false, fa1.v, false, fb1.v, (short)0, accs[1][1], false, false);
  }

  // epilogue
#pragma unroll
  for (int sm = 0; sm < 2; ++sm) {
#pragma unroll
    for (int sn = 0; sn < 2; ++sn) {
      int gcol = nBlock + wn * 32 + sn * 16 + lm;
#pragma unroll
      for (int i = 0; i < 8; ++i) {
        int grow = mBlock + wm * 32 + sm * 16 + half * 8 + i;
        if (grow >= Mreal) continue;
        float v = accs[sm][sn][i] * scale;
        if (mode == 4) {                  // head-merge: z=b*H+h, grow=t
          if (gcol >= DKn) continue;
          int b = (int)z / Hn, h = (int)z % Hn;
          Cb[((long)(b * Tn + grow)) * ldc + h * DKn + gcol] = (__bf16)v;
          continue;
        }
        if (gcol >= Nreal) continue;
        if (bias) v += bias[gcol];
        if (act == 1)      v = 0.5f * v * (1.0f + erff(v * 0.70710678118f));
        else if (act == 2) v = 1.0f / (1.0f + __expf(-v));
        if (mode == 0)      Cb[z * sC + (long)grow * ldc + gcol] = (__bf16)v;
        else if (mode == 1) Cf[z * sC + (long)grow * ldc + gcol] = v;
        else if (mode == 2) Cf[(long)grow * ldc + gcol] += v;
        else {                            // head-split proj store
          int h = gcol / DKn, dk = gcol % DKn;
          int b = grow / Tn, t = grow % Tn;
          Cb[(((long)(b * Hn + h)) * Tn + t) * (long)DKPn + dk] = (__bf16)v;
        }
      }
    }
  }
}

// ---------------------------------------------------------------------------
// LayerNorm, one wave per row (wave32). Optional additive input (pos emb),
// optional fp32 and/or bf16 outputs (row stride DPn).
// ---------------------------------------------------------------------------
__global__ void ln_kernel(const float* __restrict__ X, const float* __restrict__ add,
                          const float* __restrict__ g, const float* __restrict__ bta,
                          float* outf, __bf16* outb, int rows) {
  int row = blockIdx.x * 8 + (threadIdx.x >> 5);
  if (row >= rows) return;
  int lane = threadIdx.x & 31;
  const float* xr = X + (long)row * DPn;
  const float* ar = add ? (add + (long)(row % Tn) * Dn) : nullptr;
  float s = 0.f, s2 = 0.f;
  for (int c = lane; c < Dn; c += 32) {
    float v = xr[c];
    if (ar) v += ar[c];
    s += v; s2 += v * v;
  }
  for (int off = 16; off > 0; off >>= 1) {
    s  += __shfl_down(s, off, 32);
    s2 += __shfl_down(s2, off, 32);
  }
  s = __shfl(s, 0, 32); s2 = __shfl(s2, 0, 32);
  float mu  = s / Dn;
  float var = s2 / Dn - mu * mu;
  float inv = rsqrtf(var + 1e-12f);
  for (int c = lane; c < Dn; c += 32) {
    float v = xr[c];
    if (ar) v += ar[c];
    float y = g[c] * (v - mu) * inv + bta[c];
    if (outf) outf[(long)row * DPn + c] = y;
    if (outb) outb[(long)row * DPn + c] = (__bf16)y;
  }
}

// ---------------------------------------------------------------------------
// Row softmax with key mask: S fp32 [BH*T, T] -> P bf16. One block per row.
// ---------------------------------------------------------------------------
__global__ __launch_bounds__(256)
void softmax_kernel(const float* __restrict__ S, const int* __restrict__ mask,
                    __bf16* __restrict__ P) {
  long row = blockIdx.x;              // z*T + q
  long zz = row / Tn;
  int b = (int)(zz / Hn);
  const float* sr = S + row * Tn;
  __bf16* pr = P + row * Tn;
  const int* mrow = mask + (long)b * Tn;
  __shared__ float red[256];
  float vals[2];
  float mymax = -3.0e38f;
#pragma unroll
  for (int i = 0; i < 2; ++i) {
    int c = threadIdx.x + i * 256;
    float v = sr[c];
    if (mrow[c] == 0) v = NEGV;
    vals[i] = v;
    mymax = fmaxf(mymax, v);
  }
  red[threadIdx.x] = mymax; __syncthreads();
  for (int s = 128; s > 0; s >>= 1) {
    if (threadIdx.x < s) red[threadIdx.x] = fmaxf(red[threadIdx.x], red[threadIdx.x + s]);
    __syncthreads();
  }
  float mx = red[0]; __syncthreads();
  float mysum = 0.f;
#pragma unroll
  for (int i = 0; i < 2; ++i) { vals[i] = __expf(vals[i] - mx); mysum += vals[i]; }
  red[threadIdx.x] = mysum; __syncthreads();
  for (int s = 128; s > 0; s >>= 1) {
    if (threadIdx.x < s) red[threadIdx.x] += red[threadIdx.x + s];
    __syncthreads();
  }
  float inv = 1.0f / red[0];
#pragma unroll
  for (int i = 0; i < 2; ++i) pr[threadIdx.x + i * 256] = (__bf16)(vals[i] * inv);
}

// highway: x = g*h + (1-g)*x   (bf16 buffers, fp32 math)
__global__ void hwy_combine_kernel(__bf16* __restrict__ X, const __bf16* __restrict__ Hb,
                                   const __bf16* __restrict__ G) {
  long idx = (long)blockIdx.x * 256 + threadIdx.x;
  long tot = (long)ROWSn * Dn;
  if (idx >= tot) return;
  long r = idx / Dn; int c = (int)(idx % Dn);
  long o = r * DPn + c;
  float x = (float)X[o], h = (float)Hb[o], g = (float)G[o];
  X[o] = (__bf16)(g * h + (1.0f - g) * x);
}

// masked mean/max pooling over T: writes pools[b][slot]=mean, pools[b][slot+1]=max
__global__ void pool_kernel(const float* __restrict__ X, const int* __restrict__ mask,
                            float* __restrict__ pools, int slot) {
  int b = blockIdx.x, d = threadIdx.x;
  if (d >= Dn) return;
  const float* xb = X + (long)b * Tn * DPn;
  const int* mb = mask + (long)b * Tn;
  float s = 0.f, cnt = 0.f, mx = NEGV;
  for (int t = 0; t < Tn; ++t) {
    if (mb[t]) {
      float v = xb[(long)t * DPn + d];
      s += v; cnt += 1.0f; mx = fmaxf(mx, v);
    }
  }
  pools[((long)b * 4 + slot) * Dn + d]     = s / fmaxf(cnt, 1.0f);
  pools[((long)b * 4 + slot + 1) * Dn + d] = mx;
}

// feat [128,1824] bf16 = [qm qx am ax qm*am |qm-am| 0-pad]; rows>=16 zeroed
__global__ void feat_kernel(const float* __restrict__ pools, __bf16* __restrict__ feat) {
  long idx = (long)blockIdx.x * 256 + threadIdx.x;
  long tot = 128L * 1824;
  if (idx >= tot) return;
  int r = (int)(idx / 1824), c = (int)(idx % 1824);
  float v = 0.0f;
  if (r < Bn && c < 1800) {
    int seg = c / Dn, d = c % Dn;
    const float* pb = pools + (long)r * 4 * Dn;
    float qm = pb[d], qx = pb[Dn + d], am = pb[2 * Dn + d], ax = pb[3 * Dn + d];
    switch (seg) {
      case 0: v = qm; break;
      case 1: v = qx; break;
      case 2: v = am; break;
      case 3: v = ax; break;
      case 4: v = qm * am; break;
      default: v = fabsf(qm - am); break;
    }
  }
  feat[(long)r * 1824 + c] = (__bf16)v;
}

// ---------------------------------------------------------------------------
// Host-side parameter index bookkeeping (JAX pytree flatten: dict keys sorted)
// ---------------------------------------------------------------------------
struct LinW  { int W, b; };
struct AttnW { LinW k, o, q, v; };           // sorted: k,o,q,v
struct FFW   { LinW f1, f2; };
struct LNW   { int w, b; };
struct EncLayerW { AttnW at; FFW ff; LNW ln_in, ln_out; };
struct EncW  { EncLayerW l[4]; LNW ln; };
struct CrLayerW { AttnW cross; FFW ff; LNW ln_in, ln_mid, ln_out; AttnW self_; };
struct CrW   { CrLayerW l[2]; LNW ln; };
struct HwyW  { LinW fc, gate0, gate1, lin0, lin1; };

extern "C" void kernel_launch(void* const* d_in, const int* in_sizes, int n_in,
                              void* d_out, int out_size, void* d_ws, size_t ws_size,
                              hipStream_t stream) {
  (void)in_sizes; (void)n_in; (void)out_size; (void)ws_size;

  // ---- top-level inputs (setup_inputs insertion order) ----
  const float* q_emb = (const float*)d_in[0];
  const float* a_emb = (const float*)d_in[1];
  const int*   qmask = (const int*)d_in[2];
  const int*   amask = (const int*)d_in[3];

  // ---- params pytree flatten (keys sorted alphabetically at every dict) ----
  int cur = 4;
  auto nxt   = [&]() { return cur++; };
  auto lin   = [&]() { LinW l; l.W = nxt(); l.b = nxt(); return l; };
  auto attn  = [&]() { AttnW a; a.k = lin(); a.o = lin(); a.q = lin(); a.v = lin(); return a; };
  auto ffp   = [&]() { FFW f; f.f1 = lin(); f.f2 = lin(); return f; };
  auto lnp   = [&]() { LNW l; l.w = nxt(); l.b = nxt(); return l; };
  auto crlay = [&]() { CrLayerW c; c.cross = attn(); c.ff = ffp(); c.ln_in = lnp();
                       c.ln_mid = lnp(); c.ln_out = lnp(); c.self_ = attn(); return c; };
  auto enclay= [&]() { EncLayerW e; e.at = attn(); e.ff = ffp(); e.ln_in = lnp();
                       e.ln_out = lnp(); return e; };
  LinW clfW = lin();                                     // 'clf'
  CrW cr_aq; cr_aq.l[0] = crlay(); cr_aq.l[1] = crlay(); cr_aq.ln = lnp();   // 'cr_aq'
  CrW cr_qa; cr_qa.l[0] = crlay(); cr_qa.l[1] = crlay(); cr_qa.ln = lnp();   // 'cr_qa'
  EncW enc_a; for (int i = 0; i < 4; ++i) enc_a.l[i] = enclay(); enc_a.ln = lnp(); // 'enc_a'
  EncW enc_q; for (int i = 0; i < 4; ++i) enc_q.l[i] = enclay(); enc_q.ln = lnp(); // 'enc_q'
  HwyW hw; hw.fc = lin(); hw.gate0 = lin(); hw.gate1 = lin(); hw.lin0 = lin(); hw.lin1 = lin(); // 'highway'
  int pos_emb = nxt(); LNW pos_ln = lnp();               // 'pos': emb, ln
  LinW projW = lin();                                    // 'proj'

  // ---- workspace carve-up ----
  size_t off = 0;
  char* WSP = (char*)d_ws;
  auto carve = [&](size_t bytes) { size_t o = off; off = (off + bytes + 255) & ~(size_t)255; return WSP + o; };
  float*  Xq   = (float*) carve((size_t)ROWSn * DPn * 4);
  float*  Xa   = (float*) carve((size_t)ROWSn * DPn * 4);
  __bf16* R    = (__bf16*)carve((size_t)ROWSn * DPn * 2);
  __bf16* Xb   = (__bf16*)carve((size_t)ROWSn * DPn * 2);
  __bf16* Hb   = (__bf16*)carve((size_t)ROWSn * DPn * 2);
  __bf16* Gb   = (__bf16*)carve((size_t)ROWSn * DPn * 2);
  __bf16* Qh   = (__bf16*)carve((size_t)BHn * Tn * DKPn * 2);
  __bf16* Kh   = (__bf16*)carve((size_t)BHn * Tn * DKPn * 2);
  __bf16* Vh   = (__bf16*)carve((size_t)BHn * Tn * DKPn * 2);
  float*  S    = (float*) carve((size_t)BHn * Tn * Tn * 4);
  __bf16* P    = (__bf16*)carve((size_t)BHn * Tn * Tn * 2);
  __bf16* Mg   = (__bf16*)carve((size_t)ROWSn * DPn * 2);
  __bf16* F    = (__bf16*)carve((size_t)ROWSn * HIDn * 2);
  __bf16* Mq   = (__bf16*)carve((size_t)ROWSn * DPn * 2);
  __bf16* Ma   = (__bf16*)carve((size_t)ROWSn * DPn * 2);
  __bf16* wb   = (__bf16*)carve((size_t)1824 * 512 * 2);
  float*  pools= (float*) carve((size_t)Bn * 4 * Dn * 4);
  __bf16* feat = (__bf16*)carve((size_t)128 * 1824 * 2);
  __bf16* ProjO= (__bf16*)carve((size_t)128 * HIDn * 2);

  // ---- launch helpers ----
  auto fp = [&](int idx) { return (const float*)d_in[idx]; };

  auto gemm = [&](const __bf16* A, int lda, long sA,
                  const __bf16* Bm, int ldb, long sB, int bmode,
                  const float* bias, float* Cf, __bf16* Cb, int ldc, long sC,
                  int Kp, int Mrows, int Mreal, int Nt, int Nreal,
                  int mode, int act, float scale, int batch) {
    dim3 g(Nt, Mrows / 128, batch), blk(256);
    gemm_kernel<<<g, blk, 0, stream>>>(A, lda, sA, Bm, ldb, sB, bmode, bias,
                                       Cf, Cb, ldc, sC, Kp, Mreal, Nreal,
                                       mode, act, scale);
  };
  auto wcvt = [&](int wIdx, int K, int N, int Kp, int Np) {
    long tot = (long)Kp * Np;
    cvt_pad_kernel<<<dim3((unsigned)((tot + 255) / 256)), 256, 0, stream>>>(
        fp(wIdx), K, N, N, wb, Kp, Np, Np);
  };
  auto linear = [&](LinW w, int K, int N, const __bf16* A, int lda, int Mrows, int Mreal,
                    int mode, int act, float* Cf, __bf16* Cb, int ldc) {
    int Kp = ((K + 31) / 32) * 32, Np = ((N + 63) / 64) * 64;
    wcvt(w.W, K, N, Kp, Np);
    gemm(A, lda, 0, wb, Np, 0, 0, fp(w.b), Cf, Cb, ldc, 0,
         Kp, Mrows, Mreal, Np / 64, N, mode, act, 1.0f, 1);
  };
  auto lnrun = [&](const float* X, LNW w, float* outf, __bf16* outb, const float* add) {
    ln_kernel<<<dim3((ROWSn + 7) / 8), 256, 0, stream>>>(X, add, fp(w.w), fp(w.b),
                                                         outf, outb, ROWSn);
  };
  const float kScale = 0.11547005383792516f;   // 1/sqrt(75)
  auto mha = [&](AttnW& p, const __bf16* Rq, const __bf16* Rkv, const int* mask, float* Xres) {
    // Q/K/V projections with head-split stores
    wcvt(p.q.W, Dn, Dn, DPn, DPn);
    gemm(Rq,  DPn, 0, wb, DPn, 0, 0, fp(p.q.b), nullptr, Qh, DKPn, 0,
         DPn, ROWSn, ROWSn, DPn / 64, Dn, 3, 0, 1.0f, 1);
    wcvt(p.k.W, Dn, Dn, DPn, DPn);
    gemm(Rkv, DPn, 0, wb, DPn, 0, 0, fp(p.k.b), nullptr, Kh, DKPn, 0,
         DPn, ROWSn, ROWSn, DPn / 64, Dn, 3, 0, 1.0f, 1);
    wcvt(p.v.W, Dn, Dn, DPn, DPn);
    gemm(Rkv, DPn, 0, wb, DPn, 0, 0, fp(p.v.b), nullptr, Vh, DKPn, 0,
         DPn, ROWSn, ROWSn, DPn / 64, Dn, 3, 0, 1.0f, 1);
    // S = (Q @ K^T) / sqrt(dk), batched over B*H (async B path, bmode=1)
    gemm(Qh, DKPn, (long)Tn * DKPn, Kh, DKPn, (long)Tn * DKPn, 1, nullptr,
         S, nullptr, Tn, (long)Tn * Tn,
         DKPn, Tn, Tn, Tn / 64, Tn, 1, 0, kScale, BHn);
    // softmax(masked) -> P bf16
    softmax_kernel<<<dim3(BHn * Tn), 256, 0, stream>>>(S, mask, P);
    // O_heads = P @ V, merged directly into Mg [8192,320]
    gemm(P, Tn, (long)Tn * Tn, Vh, DKPn, (long)Tn * DKPn, 0, nullptr,
         nullptr, Mg, DPn, 0, Tn, Tn, Tn, DKPn / 64, DKPn, 4, 0, 1.0f, BHn);
    // output projection with residual add into the fp32 stream
    wcvt(p.o.W, Dn, Dn, DPn, DPn);
    gemm(Mg, DPn, 0, wb, DPn, 0, 0, fp(p.o.b), Xres, nullptr, DPn, 0,
         DPn, ROWSn, ROWSn, DPn / 64, Dn, 2, 0, 1.0f, 1);
  };
  auto ffrun = [&](FFW& f, float* Xres) {
    linear(f.f1, Dn, HIDn, R, DPn, ROWSn, ROWSn, 0, 1, nullptr, F, HIDn);      // gelu
    linear(f.f2, HIDn, Dn, F, HIDn, ROWSn, ROWSn, 2, 0, Xres, nullptr, DPn);   // +resid
  };
  auto highway = [&](const float* emb, float* Xout) {
    long tot = (long)ROWSn * DPn;
    cvt_pad_kernel<<<dim3((unsigned)((tot + 255) / 256)), 256, 0, stream>>>(
        emb, ROWSn, Dn, Dn, Xb, ROWSn, DPn, DPn);
    LinW lins[2]  = {hw.lin0, hw.lin1};
    LinW gates[2] = {hw.gate0, hw.gate1};
    for (int l = 0; l < 2; ++l) {
      linear(lins[l],  Dn, Dn, Xb, DPn, ROWSn, ROWSn, 0, 1, nullptr, Hb, DPn); // gelu
      linear(gates[l], Dn, Dn, Xb, DPn, ROWSn, ROWSn, 0, 2, nullptr, Gb, DPn); // sigmoid
      long t2 = (long)ROWSn * Dn;
      hwy_combine_kernel<<<dim3((unsigned)((t2 + 255) / 256)), 256, 0, stream>>>(Xb, Hb, Gb);
    }
    linear(hw.fc, Dn, Dn, Xb, DPn, ROWSn, ROWSn, 1, 0, Xout, nullptr, DPn);    // fp32 out
  };
  auto encode = [&](EncW& e, float* X, const int* mask, __bf16* Mem) {
    for (int l = 0; l < 4; ++l) {
      lnrun(X, e.l[l].ln_in, nullptr, R, nullptr);
      mha(e.l[l].at, R, R, mask, X);
      lnrun(X, e.l[l].ln_out, nullptr, R, nullptr);
      ffrun(e.l[l].ff, X);
    }
    lnrun(X, e.ln, X, Mem, nullptr);   // final LN in-place + bf16 memory snapshot
  };
  auto crossblock = [&](CrW& c, float* X, const __bf16* Mem,
                        const int* smask, const int* cmask) {
    for (int l = 0; l < 2; ++l) {
      lnrun(X, c.l[l].ln_in, nullptr, R, nullptr);
      mha(c.l[l].self_, R, R, smask, X);
      lnrun(X, c.l[l].ln_mid, nullptr, R, nullptr);
      mha(c.l[l].cross, R, Mem, cmask, X);
      lnrun(X, c.l[l].ln_out, nullptr, R, nullptr);
      ffrun(c.l[l].ff, X);
    }
    lnrun(X, c.ln, X, nullptr, nullptr);
  };

  // ---- forward pass ----
  {  // zero head-pad columns once (persist for the whole pass)
    long tot = (long)BHn * Tn * (DKPn - DKn);
    dim3 g((unsigned)((tot + 255) / 256));
    zero_headpad_kernel<<<g, 256, 0, stream>>>(Qh);
    zero_headpad_kernel<<<g, 256, 0, stream>>>(Kh);
    zero_headpad_kernel<<<g, 256, 0, stream>>>(Vh);
  }

  highway(q_emb, Xq);
  lnrun(Xq, pos_ln, Xq, nullptr, fp(pos_emb));   // x = LN(x + pos)
  highway(a_emb, Xa);
  lnrun(Xa, pos_ln, Xa, nullptr, fp(pos_emb));

  encode(enc_q, Xq, qmask, Mq);
  encode(enc_a, Xa, amask, Ma);

  crossblock(cr_qa, Xq, Ma, qmask, amask);
  crossblock(cr_aq, Xa, Mq, amask, qmask);

  pool_kernel<<<dim3(Bn), 320, 0, stream>>>(Xq, qmask, pools, 0);
  pool_kernel<<<dim3(Bn), 320, 0, stream>>>(Xa, amask, pools, 2);
  {
    long tot = 128L * 1824;
    feat_kernel<<<dim3((unsigned)((tot + 255) / 256)), 256, 0, stream>>>(pools, feat);
  }
  // head MLP: gelu(feat @ proj) @ clf -> d_out [16,3] fp32
  linear(projW, 1800, HIDn, feat, 1824, 128, 128, 0, 1, nullptr, ProjO, HIDn);
  linear(clfW, HIDn, 3, ProjO, HIDn, 128, Bn, 1, 0, (float*)d_out, nullptr, 3);
}
